// GVPModel_16295105921552
// MI455X (gfx1250) — matrix operand order repaired and checked by reading, not compiled, not run on hardware
//
#include <hip/hip_runtime.h>
#include <hip/hip_bf16.h>
#include <cstdint>

// ---------------------------------------------------------------------------
// GVP-GNN forward for MI455X (gfx1250). All big GEMMs via v_wmma_f32_16x16x32_bf16.
// ---------------------------------------------------------------------------

#define N_NODES 30000
#define N_EDGES 300000
#define N_GRAPHS 64
#define NSP 112            // padded node-scalar stride (NS=100)
#define VST 48             // node vector stride (16*3)
#define EPSN 1e-8f

typedef __bf16 bf16x16 __attribute__((ext_vector_type(16)));
typedef float  f32x8   __attribute__((ext_vector_type(8)));

__device__ __forceinline__ unsigned short f2bf(float x) {
  unsigned int u = __float_as_uint(x);
  u += 0x7FFFu + ((u >> 16) & 1u);          // round-to-nearest-even
  return (unsigned short)(u >> 16);
}

// One 16x16x32 bf16 WMMA step. A from LDS (row-major bf16, rowStride elems),
// B from packed global weights: [(kChunk*nTiles+nTile)][half][16 cols][16 k] bf16.
__device__ __forceinline__ f32x8 wmma_tile(const unsigned short* Xl, int rowStride, int kChunk,
                                           const unsigned short* Bp, int nTiles, int nTile,
                                           f32x8 acc) {
  int lane = threadIdx.x & 31;
  int row  = lane & 15;
  int hi   = lane >> 4;
  union { unsigned int u[8]; bf16x16 v; } A, B;
  const unsigned int* pa = (const unsigned int*)(Xl + row * rowStride + kChunk * 32 + hi * 16);
#pragma unroll
  for (int i = 0; i < 4; ++i) { A.u[i] = pa[i]; A.u[4 + i] = pa[4 + i]; }
  const unsigned int* pb = (const unsigned int*)
      (Bp + (((size_t)(kChunk * nTiles + nTile) * 2 + hi) * 16 + row) * 16);
#pragma unroll
  for (int i = 0; i < 8; ++i) B.u[i] = pb[i];
  return __builtin_amdgcn_wmma_f32_16x16x32_bf16(false, A.v, false, B.v, (short)0, acc,
                                                 false, false);
}

// ---------------------------------------------------------------------------
// Utility kernels
// ---------------------------------------------------------------------------
__global__ void fill0_kernel(float* p, long n) {
  long i = (long)blockIdx.x * blockDim.x + threadIdx.x;
  long stride = (long)gridDim.x * blockDim.x;
  for (; i < n; i += stride) p[i] = 0.f;
}

__global__ void degree_kernel(const int* __restrict__ ids, float* __restrict__ cnt, int n) {
  int i = blockIdx.x * blockDim.x + threadIdx.x;
  if (i < n) atomicAdd(&cnt[ids[i]], 1.f);
}

// Pack fp32 [Kact x Nact] row-major -> bf16 WMMA B-fragment tiles (zero padded).
__global__ void pack_b_kernel(const float* __restrict__ W, int Kact, int Nact,
                              int Kpad, int Npad, unsigned short* __restrict__ out) {
  int nT = Npad >> 4;
  int total = Kpad * Npad;
  for (int o = blockIdx.x * blockDim.x + threadIdx.x; o < total;
       o += gridDim.x * blockDim.x) {
    int c2 = o & 511, blk = o >> 9;
    int nTile = blk % nT, kChunk = blk / nT;
    int kk = c2 & 15, n = (c2 >> 4) & 15, h = c2 >> 8;
    int K = kChunk * 32 + h * 16 + kk;
    int N = nTile * 16 + n;
    float v = (K < Kact && N < Nact) ? W[(size_t)K * Nact + N] : 0.f;
    out[o] = f2bf(v);
  }
}

// ---------------------------------------------------------------------------
// Embeddings. LN over a single scalar collapses to its bias; v-LN normalizes |v|.
// ---------------------------------------------------------------------------
__global__ void embed_node_s_kernel(const float* __restrict__ nv, const float* __restrict__ lnb,
                                    const float* __restrict__ wh, const float* __restrict__ wsw,
                                    const float* __restrict__ wsb, float* __restrict__ sOut) {
  long gid = (long)blockIdx.x * blockDim.x + threadIdx.x;
  if (gid >= (long)N_NODES * 100) return;
  int node = (int)(gid / 100), o = (int)(gid - (long)node * 100);
  float vx = nv[(size_t)node * 3], vy = nv[(size_t)node * 3 + 1], vz = nv[(size_t)node * 3 + 2];
  float nsq = vx * vx + vy * vy + vz * vz;
  float inv = rsqrtf(fmaxf(nsq, EPSN));
  float vv = nsq * inv * inv;                  // |v_ln|^2
  float acc = wsb[o] + lnb[0] * wsw[o];
#pragma unroll
  for (int h = 0; h < 16; ++h) {
    float w = wh[h];
    acc += sqrtf(fmaxf(w * w * vv, EPSN)) * wsw[(1 + h) * 100 + o];
  }
  sOut[(size_t)node * NSP + o] = acc;
}

__global__ void embed_node_v_kernel(const float* __restrict__ nv, const float* __restrict__ sBuf,
                                    const float* __restrict__ wh, const float* __restrict__ wv,
                                    const float* __restrict__ wsvw, const float* __restrict__ wsvb,
                                    float* __restrict__ vOut) {
  long gid = (long)blockIdx.x * blockDim.x + threadIdx.x;
  if (gid >= (long)N_NODES * 16) return;
  int node = (int)(gid >> 4), o = (int)(gid & 15);
  float g = wsvb[o];
  for (int s = 0; s < 100; ++s) g += sBuf[(size_t)node * NSP + s] * wsvw[s * 16 + o];
  g = 1.f / (1.f + __expf(-g));
  float k = 0.f;
#pragma unroll
  for (int h = 0; h < 16; ++h) k += wh[h] * wv[h * 16 + o];
  float vx = nv[(size_t)node * 3], vy = nv[(size_t)node * 3 + 1], vz = nv[(size_t)node * 3 + 2];
  float nsq = vx * vx + vy * vy + vz * vz;
  float inv = rsqrtf(fmaxf(nsq, EPSN));
  float c = g * k * inv;
  vOut[(size_t)node * VST + o * 3 + 0] = c * vx;
  vOut[(size_t)node * VST + o * 3 + 1] = c * vy;
  vOut[(size_t)node * VST + o * 3 + 2] = c * vz;
}

__global__ void embed_edge_s_kernel(const float* __restrict__ evr, const float* __restrict__ lnb,
                                    const float* __restrict__ wh, const float* __restrict__ wsw,
                                    const float* __restrict__ wsb, float* __restrict__ esOut) {
  long gid = (long)blockIdx.x * blockDim.x + threadIdx.x;
  if (gid >= (long)N_EDGES * 32) return;
  int e = (int)(gid >> 5), o = (int)(gid & 31);
  float vx = evr[(size_t)e * 3], vy = evr[(size_t)e * 3 + 1], vz = evr[(size_t)e * 3 + 2];
  float nsq = vx * vx + vy * vy + vz * vz;
  float inv = rsqrtf(fmaxf(nsq, EPSN));
  float vv = nsq * inv * inv;
  float w0 = wh[0];
  float vn0 = sqrtf(fmaxf(w0 * w0 * vv, EPSN));
  esOut[(size_t)e * 32 + o] = wsb[o] + lnb[0] * wsw[o] + vn0 * wsw[32 + o];
}

__global__ void embed_edge_v_kernel(const float* __restrict__ evr, const float* __restrict__ esB,
                                    const float* __restrict__ wh, const float* __restrict__ wv,
                                    const float* __restrict__ wsvw, const float* __restrict__ wsvb,
                                    float* __restrict__ evOut) {
  int e = blockIdx.x * blockDim.x + threadIdx.x;
  if (e >= N_EDGES) return;
  float g = wsvb[0];
  for (int o = 0; o < 32; ++o) g += esB[(size_t)e * 32 + o] * wsvw[o];
  g = 1.f / (1.f + __expf(-g));
  float k = wh[0] * wv[0];
  float vx = evr[(size_t)e * 3], vy = evr[(size_t)e * 3 + 1], vz = evr[(size_t)e * 3 + 2];
  float nsq = vx * vx + vy * vy + vz * vz;
  float inv = rsqrtf(fmaxf(nsq, EPSN));
  float c = g * k * inv;
  evOut[(size_t)e * 3 + 0] = c * vx;
  evOut[(size_t)e * 3 + 1] = c * vy;
  evOut[(size_t)e * 3 + 2] = c * vz;
}

// ---------------------------------------------------------------------------
// Fused 3-GVP message kernel: 16 edges / WG, 4 waves, WMMA s-GEMMs, atomic scatter.
// X1: [16 x 288] bf16 (ms 232 | vn 33 | pad). X2/X3 reuse cols [0,128).
// ---------------------------------------------------------------------------
__global__ __launch_bounds__(128) void msg_kernel(
    const float* __restrict__ sN, const float* __restrict__ vN,
    const float* __restrict__ esB, const float* __restrict__ evB,
    const int* __restrict__ srcI, const int* __restrict__ dstI,
    const unsigned short* __restrict__ W1p, const unsigned short* __restrict__ W2p,
    const unsigned short* __restrict__ W3p,
    const float* __restrict__ wh1, const float* __restrict__ b1, const float* __restrict__ wv1,
    const float* __restrict__ wsv1, const float* __restrict__ wsvb1,
    const float* __restrict__ wh2, const float* __restrict__ b2, const float* __restrict__ wv2,
    const float* __restrict__ wsv2, const float* __restrict__ wsvb2,
    const float* __restrict__ wh3, const float* __restrict__ b3, const float* __restrict__ wv3,
    const float* __restrict__ wsv3, const float* __restrict__ wsvb3,
    float* __restrict__ dsS, float* __restrict__ dvS) {
  __shared__ unsigned short Xa[16 * 288];
  __shared__ float Sout[16 * NSP];
  __shared__ float Vh[16 * 99];
  __shared__ float Vc[16 * 99];
  __shared__ float Gt[16 * 16];
  __shared__ int eS[16], eD[16];

  int tid = threadIdx.x, wave = tid >> 5, lane = tid & 31;
  int e0 = blockIdx.x * 16;

  if (tid < 16) { eS[tid] = srcI[e0 + tid]; eD[tid] = dstI[e0 + tid]; }
  { unsigned int* xu = (unsigned int*)Xa;
    for (int i = tid; i < 16 * 288 / 2; i += 128) xu[i] = 0u; }
  __syncthreads();

  // ms = [s_src | es | s_dst] -> bf16
  for (int idx = tid; idx < 16 * 232; idx += 128) {
    int i = idx / 232, k = idx - i * 232;
    float v;
    if (k < 100)       v = sN[(size_t)eS[i] * NSP + k];
    else if (k < 132)  v = esB[(size_t)(e0 + i) * 32 + (k - 100)];
    else               v = sN[(size_t)eD[i] * NSP + (k - 132)];
    Xa[i * 288 + k] = f2bf(v);
  }
  // mv = [v_src | ev | v_dst]
  for (int idx = tid; idx < 16 * 99; idx += 128) {
    int i = idx / 99, r = idx - i * 99, j = r / 3, c = r - j * 3;
    float v;
    if (j < 16)        v = vN[(size_t)eS[i] * VST + j * 3 + c];
    else if (j == 16)  v = evB[(size_t)(e0 + i) * 3 + c];
    else               v = vN[(size_t)eD[i] * VST + (j - 17) * 3 + c];
    Vc[idx] = v;
  }
  __syncthreads();

  // vh1 = wh1^T . mv  (h=33)
  for (int idx = tid; idx < 16 * 99; idx += 128) {
    int i = idx / 99, r = idx - i * 99, h = r / 3, c = r - h * 3;
    float acc = 0.f;
    for (int j = 0; j < 33; ++j) acc += Vc[i * 99 + j * 3 + c] * wh1[j * 33 + h];
    Vh[idx] = acc;
  }
  __syncthreads();
  // vn1 -> X cols 232..264
  for (int idx = tid; idx < 16 * 33; idx += 128) {
    int i = idx / 33, h = idx - i * 33;
    float a = Vh[i * 99 + h * 3], b_ = Vh[i * 99 + h * 3 + 1], c_ = Vh[i * 99 + h * 3 + 2];
    Xa[i * 288 + 232 + h] = f2bf(sqrtf(fmaxf(a * a + b_ * b_ + c_ * c_, EPSN)));
  }
  __syncthreads();

  // GEMM1: 7 N-tiles x 9 K-chunks (K=288)
  for (int t = wave; t < 7; t += 4) {
    f32x8 acc = {0.f, 0.f, 0.f, 0.f, 0.f, 0.f, 0.f, 0.f};
    for (int k = 0; k < 9; ++k) acc = wmma_tile(Xa, 288, k, W1p, 7, t, acc);
    int n = t * 16 + (lane & 15);
    float bias = (n < 100) ? b1[n] : 0.f;
    int mb = (lane >> 4) * 8;
    for (int r = 0; r < 8; ++r) Sout[(mb + r) * NSP + n] = acc[r] + bias;
  }
  __syncthreads();

  const float* wvA[3]   = {wv1, wv2, wv3};
  const float* wsvA[3]  = {wsv1, wsv2, wsv3};
  const float* wsvbA[3] = {wsvb1, wsvb2, wsvb3};
  const float* whN[2]   = {wh2, wh3};
  const float* bN[2]    = {b2, b3};
  const unsigned short* WpN[2] = {W2p, W3p};

  for (int g = 0; g < 3; ++g) {
    int H = (g == 0) ? 33 : 16;
    // gate from pre-activation s_out
    for (int idx = tid; idx < 256; idx += 128) {
      int i = idx >> 4, o = idx & 15;
      float acc = wsvbA[g][o];
      for (int s = 0; s < 100; ++s) acc += Sout[i * NSP + s] * wsvA[g][s * 16 + o];
      Gt[idx] = 1.f / (1.f + __expf(-acc));
    }
    __syncthreads();
    // v_out -> Vc (16 channels)
    for (int idx = tid; idx < 16 * 48; idx += 128) {
      int i = idx / 48, r = idx - i * 48, o = r / 3, c = r - o * 3;
      float acc = 0.f;
      for (int h = 0; h < H; ++h) acc += Vh[i * 99 + h * 3 + c] * wvA[g][h * 16 + o];
      Vc[i * 99 + r] = acc * Gt[i * 16 + o];
    }
    if (g == 2) break;
    __syncthreads();
    // X_{g+1}: cols 0..99 relu(s_out), 116..127 zero
    for (int idx = tid; idx < 16 * 128; idx += 128) {
      int i = idx >> 7, k = idx & 127;
      if (k < 100)       Xa[i * 288 + k] = f2bf(fmaxf(Sout[i * NSP + k], 0.f));
      else if (k >= 116) Xa[i * 288 + k] = 0;
    }
    __syncthreads();
    // vh = wh^T . v  (16x16)
    for (int idx = tid; idx < 16 * 48; idx += 128) {
      int i = idx / 48, r = idx - i * 48, h = r / 3, c = r - h * 3;
      float acc = 0.f;
      for (int j = 0; j < 16; ++j) acc += Vc[i * 99 + j * 3 + c] * whN[g][j * 16 + h];
      Vh[i * 99 + r] = acc;
    }
    __syncthreads();
    // vn -> cols 100..115
    for (int idx = tid; idx < 256; idx += 128) {
      int i = idx >> 4, h = idx & 15;
      float a = Vh[i * 99 + h * 3], b_ = Vh[i * 99 + h * 3 + 1], c_ = Vh[i * 99 + h * 3 + 2];
      Xa[i * 288 + 100 + h] = f2bf(sqrtf(fmaxf(a * a + b_ * b_ + c_ * c_, EPSN)));
    }
    __syncthreads();
    // GEMM: 7 N-tiles x 4 K-chunks (K=128)
    for (int t = wave; t < 7; t += 4) {
      f32x8 acc = {0.f, 0.f, 0.f, 0.f, 0.f, 0.f, 0.f, 0.f};
      for (int k = 0; k < 4; ++k) acc = wmma_tile(Xa, 288, k, WpN[g], 7, t, acc);
      int n = t * 16 + (lane & 15);
      float bias = (n < 100) ? bN[g][n] : 0.f;
      int mb = (lane >> 4) * 8;
      for (int r = 0; r < 8; ++r) Sout[(mb + r) * NSP + n] = acc[r] + bias;
    }
    __syncthreads();
  }
  __syncthreads();

  // scatter-add (segment sum; mean applied in node kernel)
  for (int idx = tid; idx < 16 * 100; idx += 128) {
    int i = idx / 100, k = idx - i * 100;
    atomicAdd(&dsS[(size_t)eD[i] * NSP + k], Sout[i * NSP + k]);
  }
  for (int idx = tid; idx < 16 * 48; idx += 128) {
    int i = idx / 48, r = idx - i * 48;
    atomicAdd(&dvS[(size_t)eD[i] * VST + r], Vc[i * 99 + r]);
  }
}

// ---------------------------------------------------------------------------
// Fused node update: residual + LN0 + FF(GVP 100,16->400,32 relu; 400,32->100,16)
// + residual + LN1.   16 nodes / WG.
// ---------------------------------------------------------------------------
__global__ __launch_bounds__(128) void node_kernel(
    float* __restrict__ sN, float* __restrict__ vN,
    const float* __restrict__ dsS, const float* __restrict__ dvS, const float* __restrict__ cnt,
    const float* __restrict__ n0w, const float* __restrict__ n0b,
    const float* __restrict__ n1w, const float* __restrict__ n1b,
    const unsigned short* __restrict__ Wf0p, const float* __restrict__ whF0,
    const float* __restrict__ bF0, const float* __restrict__ wvF0,
    const float* __restrict__ wsvF0, const float* __restrict__ wsvbF0,
    const unsigned short* __restrict__ Wf1p, const float* __restrict__ whF1,
    const float* __restrict__ bF1, const float* __restrict__ wvF1,
    const float* __restrict__ wsvF1, const float* __restrict__ wsvbF1) {
  __shared__ float A[16 * 416];          // Sout400; later aliased: ds2 @0, v_hidden @2048
  __shared__ unsigned short X[16 * 448];
  __shared__ float Sc[16 * NSP];
  __shared__ float Vc[16 * 48];
  __shared__ float Vh[16 * 96];
  __shared__ float G32[16 * 32];
  __shared__ float Red[64];

  int tid = threadIdx.x, wave = tid >> 5, lane = tid & 31;
  int i0 = blockIdx.x * 16;

  // s residual (mean aggregation) -------------------------------------------
  for (int idx = tid; idx < 16 * 100; idx += 128) {
    int i = idx / 100, k = idx - i * 100;
    float c = fmaxf(cnt[i0 + i], 1.f);
    Sc[i * NSP + k] = sN[(size_t)(i0 + i) * NSP + k] + dsS[(size_t)(i0 + i) * NSP + k] / c;
  }
  __syncthreads();
  if (tid < 16) {
    float mu = 0.f;
    for (int k = 0; k < 100; ++k) mu += Sc[tid * NSP + k];
    mu *= 0.01f;
    float var = 0.f;
    for (int k = 0; k < 100; ++k) { float d = Sc[tid * NSP + k] - mu; var += d * d; }
    Red[tid] = mu; Red[16 + tid] = rsqrtf(var * 0.01f + 1e-5f);
  }
  __syncthreads();
  for (int idx = tid; idx < 16 * 100; idx += 128) {
    int i = idx / 100, k = idx - i * 100;
    Sc[i * NSP + k] = (Sc[i * NSP + k] - Red[i]) * Red[16 + i] * n0w[k] + n0b[k];
  }
  for (int idx = tid; idx < 16 * 48; idx += 128) {
    int i = idx / 48, r = idx - i * 48;
    float c = fmaxf(cnt[i0 + i], 1.f);
    Vc[idx] = vN[(size_t)(i0 + i) * VST + r] + dvS[(size_t)(i0 + i) * VST + r] / c;
  }
  __syncthreads();
  if (tid < 16) {
    float acc = 0.f;
    for (int j = 0; j < 16; ++j) {
      float a = Vc[tid * 48 + j * 3], b_ = Vc[tid * 48 + j * 3 + 1], c_ = Vc[tid * 48 + j * 3 + 2];
      acc += fmaxf(a * a + b_ * b_ + c_ * c_, EPSN);
    }
    Red[32 + tid] = rsqrtf(acc * (1.f / 16.f));
  }
  __syncthreads();
  for (int idx = tid; idx < 16 * 48; idx += 128) { int i = idx / 48; Vc[idx] *= Red[32 + i]; }
  __syncthreads();

  // FF0 ---------------------------------------------------------------------
  for (int idx = tid; idx < 16 * 96; idx += 128) {      // vh (h=32)
    int i = idx / 96, r = idx - i * 96, h = r / 3, c = r - h * 3;
    float acc = 0.f;
    for (int j = 0; j < 16; ++j) acc += Vc[i * 48 + j * 3 + c] * whF0[j * 32 + h];
    Vh[idx] = acc;
  }
  for (int idx = tid; idx < 16 * 160; idx += 128) {     // X cols 0..99 + zero 132..159
    int i = idx / 160, k = idx - i * 160;
    if (k < 100)       X[i * 448 + k] = f2bf(Sc[i * NSP + k]);
    else if (k >= 132) X[i * 448 + k] = 0;
  }
  __syncthreads();
  for (int idx = tid; idx < 16 * 32; idx += 128) {      // vn -> cols 100..131
    int i = idx >> 5, h = idx & 31;
    float a = Vh[i * 96 + h * 3], b_ = Vh[i * 96 + h * 3 + 1], c_ = Vh[i * 96 + h * 3 + 2];
    X[i * 448 + 100 + h] = f2bf(sqrtf(fmaxf(a * a + b_ * b_ + c_ * c_, EPSN)));
  }
  __syncthreads();
  for (int t = wave; t < 26; t += 4) {                  // GEMM ff0: N=416, K=160
    f32x8 acc = {0.f, 0.f, 0.f, 0.f, 0.f, 0.f, 0.f, 0.f};
    for (int k = 0; k < 5; ++k) acc = wmma_tile(X, 448, k, Wf0p, 26, t, acc);
    int n = t * 16 + (lane & 15);
    float bias = (n < 400) ? bF0[n] : 0.f;
    int mb = (lane >> 4) * 8;
    for (int r = 0; r < 8; ++r) A[(mb + r) * 416 + n] = acc[r] + bias;
  }
  __syncthreads();
  for (int idx = tid; idx < 16 * 32; idx += 128) {      // gate32 (pre-activation)
    int i = idx >> 5, o = idx & 31;
    float acc = wsvbF0[o];
    for (int s = 0; s < 400; ++s) acc += A[i * 416 + s] * wsvF0[s * 32 + o];
    G32[idx] = 1.f / (1.f + __expf(-acc));
  }
  for (int idx = tid; idx < 16 * 448; idx += 128) {     // X2: relu(s400) | pad
    int i = idx / 448, k = idx - i * 448;
    if (k < 400)       X[idx] = f2bf(fmaxf(A[i * 416 + k], 0.f));
    else if (k >= 432) X[idx] = 0;
  }
  __syncthreads();

  float* VHid = &A[2048];                               // A free now
  for (int idx = tid; idx < 16 * 96; idx += 128) {      // v_hidden = gate*(wv^T vh)
    int i = idx / 96, r = idx - i * 96, o = r / 3, c = r - o * 3;
    float acc = 0.f;
    for (int h = 0; h < 32; ++h) acc += Vh[i * 96 + h * 3 + c] * wvF0[h * 32 + o];
    VHid[idx] = acc * G32[i * 32 + o];
  }
  __syncthreads();
  for (int idx = tid; idx < 16 * 96; idx += 128) {      // vh2 = whF1^T v_hidden
    int i = idx / 96, r = idx - i * 96, h = r / 3, c = r - h * 3;
    float acc = 0.f;
    for (int j = 0; j < 32; ++j) acc += VHid[i * 96 + j * 3 + c] * whF1[j * 32 + h];
    Vh[idx] = acc;
  }
  __syncthreads();
  for (int idx = tid; idx < 16 * 32; idx += 128) {      // vn2 -> cols 400..431
    int i = idx >> 5, h = idx & 31;
    float a = Vh[i * 96 + h * 3], b_ = Vh[i * 96 + h * 3 + 1], c_ = Vh[i * 96 + h * 3 + 2];
    X[i * 448 + 400 + h] = f2bf(sqrtf(fmaxf(a * a + b_ * b_ + c_ * c_, EPSN)));
  }
  __syncthreads();
  float* ds2 = A;                                       // [16 x 112], below VHid
  for (int t = wave; t < 7; t += 4) {                   // GEMM ff1: N=112, K=448
    f32x8 acc = {0.f, 0.f, 0.f, 0.f, 0.f, 0.f, 0.f, 0.f};
    for (int k = 0; k < 14; ++k) acc = wmma_tile(X, 448, k, Wf1p, 7, t, acc);
    int n = t * 16 + (lane & 15);
    float bias = (n < 100) ? bF1[n] : 0.f;
    int mb = (lane >> 4) * 8;
    for (int r = 0; r < 8; ++r) ds2[(mb + r) * NSP + n] = acc[r] + bias;
  }
  __syncthreads();
  for (int idx = tid; idx < 16 * 16; idx += 128) {      // gate16
    int i = idx >> 4, o = idx & 15;
    float acc = wsvbF1[o];
    for (int s = 0; s < 100; ++s) acc += ds2[i * NSP + s] * wsvF1[s * 16 + o];
    G32[idx] = 1.f / (1.f + __expf(-acc));
  }
  __syncthreads();
  for (int idx = tid; idx < 16 * 100; idx += 128) {     // s residual 2
    int i = idx / 100, k = idx - i * 100;
    Sc[i * NSP + k] += ds2[i * NSP + k];
  }
  for (int idx = tid; idx < 16 * 48; idx += 128) {      // v residual 2
    int i = idx / 48, r = idx - i * 48, o = r / 3, c = r - o * 3;
    float acc = 0.f;
    for (int h = 0; h < 32; ++h) acc += Vh[i * 96 + h * 3 + c] * wvF1[h * 16 + o];
    Vc[idx] += acc * G32[i * 16 + o];
  }
  __syncthreads();
  if (tid < 16) {                                       // LN1 stats (s)
    float mu = 0.f;
    for (int k = 0; k < 100; ++k) mu += Sc[tid * NSP + k];
    mu *= 0.01f;
    float var = 0.f;
    for (int k = 0; k < 100; ++k) { float d = Sc[tid * NSP + k] - mu; var += d * d; }
    Red[tid] = mu; Red[16 + tid] = rsqrtf(var * 0.01f + 1e-5f);
  } else if (tid < 32) {                                // LN1 stats (v)
    int t = tid - 16;
    float acc = 0.f;
    for (int j = 0; j < 16; ++j) {
      float a = Vc[t * 48 + j * 3], b_ = Vc[t * 48 + j * 3 + 1], c_ = Vc[t * 48 + j * 3 + 2];
      acc += fmaxf(a * a + b_ * b_ + c_ * c_, EPSN);
    }
    Red[32 + t] = rsqrtf(acc * (1.f / 16.f));
  }
  __syncthreads();
  for (int idx = tid; idx < 16 * 100; idx += 128) {
    int i = idx / 100, k = idx - i * 100;
    sN[(size_t)(i0 + i) * NSP + k] = (Sc[i * NSP + k] - Red[i]) * Red[16 + i] * n1w[k] + n1b[k];
  }
  for (int idx = tid; idx < 16 * 48; idx += 128) {
    int i = idx / 48;
    vN[(size_t)(i0 + i) * VST + (idx - i * 48)] = Vc[idx] * Red[32 + i];
  }
}

// ---------------------------------------------------------------------------
// Output GVP (100,16 -> 1) + graph pooling
// ---------------------------------------------------------------------------
__global__ void out_kernel(const float* __restrict__ sN, const float* __restrict__ vN,
                           const int* __restrict__ batch, const float* __restrict__ who,
                           const float* __restrict__ wso, const float* __restrict__ bso,
                           float* __restrict__ gsum) {
  int node = blockIdx.x * blockDim.x + threadIdx.x;
  if (node >= N_NODES) return;
  float vn2[16];
#pragma unroll
  for (int h = 0; h < 16; ++h) vn2[h] = 0.f;
  for (int c = 0; c < 3; ++c) {
#pragma unroll
    for (int h = 0; h < 16; ++h) {
      float acc = 0.f;
      for (int j = 0; j < 16; ++j) acc += vN[(size_t)node * VST + j * 3 + c] * who[j * 16 + h];
      vn2[h] += acc * acc;
    }
  }
  float out = bso[0];
  for (int s = 0; s < 100; ++s) out += sN[(size_t)node * NSP + s] * wso[s];
#pragma unroll
  for (int h = 0; h < 16; ++h) out += sqrtf(fmaxf(vn2[h], EPSN)) * wso[100 + h];
  atomicAdd(&gsum[batch[node]], out);
}

__global__ void finalize_kernel(const float* __restrict__ gsum, const float* __restrict__ gcnt,
                                float* __restrict__ out) {
  int g = threadIdx.x;
  if (g < N_GRAPHS) out[g] = gsum[g] / fmaxf(gcnt[g], 1.f);
}

// ---------------------------------------------------------------------------
extern "C" void kernel_launch(void* const* d_in, const int* in_sizes, int n_in,
                              void* d_out, int out_size, void* d_ws, size_t ws_size,
                              hipStream_t stream) {
  (void)in_sizes; (void)n_in; (void)out_size; (void)ws_size;
  auto F = [&](int i) { return (const float*)d_in[i]; };
  const int* edge_index = (const int*)d_in[89];
  const int* src = edge_index;
  const int* dst = edge_index + N_EDGES;
  const int* batch = (const int*)d_in[92];
  const float* node_v = F(88);
  const float* edge_v = F(91);

  size_t off = 0;
  auto alloc = [&](size_t bytes) -> char* {
    off = (off + 255) & ~(size_t)255;
    char* p = (char*)d_ws + off;
    off += bytes;
    return p;
  };
  float* s_buf  = (float*)alloc((size_t)N_NODES * NSP * 4);
  float* v_buf  = (float*)alloc((size_t)N_NODES * VST * 4);
  float* ds_sum = (float*)alloc((size_t)N_NODES * NSP * 4);
  float* dv_sum = (float*)alloc((size_t)N_NODES * VST * 4);
  float* cnt    = (float*)alloc((size_t)N_NODES * 4);
  float* es_buf = (float*)alloc((size_t)N_EDGES * 32 * 4);
  float* ev_buf = (float*)alloc((size_t)N_EDGES * 3 * 4);
  float* gsum   = (float*)alloc(N_GRAPHS * 4);
  float* gcnt   = (float*)alloc(N_GRAPHS * 4);
  unsigned short* packW[2][5];
  const size_t szW1 = (size_t)9 * 7 * 512 * 2,  szW23 = (size_t)4 * 7 * 512 * 2;
  const size_t szF0 = (size_t)5 * 26 * 512 * 2, szF1  = (size_t)14 * 7 * 512 * 2;
  for (int L = 0; L < 2; ++L) {
    packW[L][0] = (unsigned short*)alloc(szW1);
    packW[L][1] = (unsigned short*)alloc(szW23);
    packW[L][2] = (unsigned short*)alloc(szW23);
    packW[L][3] = (unsigned short*)alloc(szF0);
    packW[L][4] = (unsigned short*)alloc(szF1);
  }

  // counts
  fill0_kernel<<<64, 256, 0, stream>>>(cnt, N_NODES);
  fill0_kernel<<<1, 64, 0, stream>>>(gsum, N_GRAPHS);
  fill0_kernel<<<1, 64, 0, stream>>>(gcnt, N_GRAPHS);
  degree_kernel<<<(N_EDGES + 255) / 256, 256, 0, stream>>>(dst, cnt, N_EDGES);
  degree_kernel<<<(N_NODES + 255) / 256, 256, 0, stream>>>(batch, gcnt, N_NODES);

  // pack WMMA B weights
  for (int L = 0; L < 2; ++L) {
    int b = 16 + 34 * L;
    pack_b_kernel<<<128, 256, 0, stream>>>(F(b + 1),  265, 100, 288, 112, packW[L][0]);
    pack_b_kernel<<<128, 256, 0, stream>>>(F(b + 7),  116, 100, 128, 112, packW[L][1]);
    pack_b_kernel<<<128, 256, 0, stream>>>(F(b + 13), 116, 100, 128, 112, packW[L][2]);
    pack_b_kernel<<<256, 256, 0, stream>>>(F(b + 19), 132, 400, 160, 416, packW[L][3]);
    pack_b_kernel<<<256, 256, 0, stream>>>(F(b + 25), 432, 100, 448, 112, packW[L][4]);
  }

  // embeddings
  embed_node_s_kernel<<<(N_NODES * 100 + 255) / 256, 256, 0, stream>>>(
      node_v, F(1), F(2), F(3), F(4), s_buf);
  embed_node_v_kernel<<<(N_NODES * 16 + 255) / 256, 256, 0, stream>>>(
      node_v, s_buf, F(2), F(5), F(6), F(7), v_buf);
  embed_edge_s_kernel<<<(N_EDGES * 32 + 255) / 256, 256, 0, stream>>>(
      edge_v, F(9), F(10), F(11), F(12), es_buf);
  embed_edge_v_kernel<<<(N_EDGES + 255) / 256, 256, 0, stream>>>(
      edge_v, es_buf, F(10), F(13), F(14), F(15), ev_buf);

  // conv layers
  for (int L = 0; L < 2; ++L) {
    int b = 16 + 34 * L;
    fill0_kernel<<<4096, 256, 0, stream>>>(ds_sum, (long)N_NODES * NSP);
    fill0_kernel<<<2048, 256, 0, stream>>>(dv_sum, (long)N_NODES * VST);
    msg_kernel<<<N_EDGES / 16, 128, 0, stream>>>(
        s_buf, v_buf, es_buf, ev_buf, src, dst,
        packW[L][0], packW[L][1], packW[L][2],
        F(b + 0),  F(b + 2),  F(b + 3),  F(b + 4),  F(b + 5),
        F(b + 6),  F(b + 8),  F(b + 9),  F(b + 10), F(b + 11),
        F(b + 12), F(b + 14), F(b + 15), F(b + 16), F(b + 17),
        ds_sum, dv_sum);
    node_kernel<<<N_NODES / 16, 128, 0, stream>>>(
        s_buf, v_buf, ds_sum, dv_sum, cnt,
        F(b + 30), F(b + 31), F(b + 32), F(b + 33),
        packW[L][3], F(b + 18), F(b + 20), F(b + 21), F(b + 22), F(b + 23),
        packW[L][4], F(b + 24), F(b + 26), F(b + 27), F(b + 28), F(b + 29));
  }

  // readout
  out_kernel<<<(N_NODES + 255) / 256, 256, 0, stream>>>(
      s_buf, v_buf, batch, F(84), F(85), F(86), gsum);
  finalize_kernel<<<1, 64, 0, stream>>>(gsum, gcnt, (float*)d_out);
}
